// MultiheadAttention_65652870087349
// MI455X (gfx1250) — matrix operand order repaired and checked by reading, not compiled, not run on hardware
//
#include <hip/hip_runtime.h>
#include <hip/hip_bf16.h>

typedef __bf16 bf16_t;
typedef __attribute__((ext_vector_type(16))) __bf16 v16bf;
typedef __attribute__((ext_vector_type(8)))  __bf16 v8bf;
typedef __attribute__((ext_vector_type(8)))  float  v8f;

#define EMBED  1024
#define NHEADS 16
#define HDIM   64
#define BB     4
#define TT     2048
#define NT     (BB * TT)        // 8192 token rows
#define E3     (3 * EMBED)      // 3072

// ---------------------------------------------------------------------------
// WMMA fragment loaders (CDNA5 16-bit layouts, CDNA5 ISA §7.12.2)
// A 16x32 : lane -> row M=lane&15, K = {kb8..kb8+7, 16+kb8..+7}, kb8=8*(lane>>4)
// B 32x16 : lane -> col N=lane&15, K = {kb8..kb8+7, 16+kb8..+7}, kb8=8*(lane>>4)
// C/D f32 : lane -> N=lane&15, VGPR r -> M = 8*(lane>>4) + r
// ---------------------------------------------------------------------------
__device__ __forceinline__ v16bf load_a_frag(const bf16_t* __restrict__ p, int stride) {
  int lane = threadIdx.x & 31;
  const bf16_t* base = p + (lane & 15) * stride + ((lane >> 4) << 3);
  v8bf lo = *(const v8bf*)(base);
  v8bf hi = *(const v8bf*)(base + 16);
  v16bf out;
#pragma unroll
  for (int i = 0; i < 8; ++i) { out[i] = lo[i]; out[i + 8] = hi[i]; }
  return out;
}

__device__ __forceinline__ v16bf load_b_frag(const bf16_t* __restrict__ p, int stride) {
  int lane = threadIdx.x & 31;
  const bf16_t* base = p + (lane & 15) * stride + ((lane >> 4) << 3);
  v8bf lo = *(const v8bf*)(base);
  v8bf hi = *(const v8bf*)(base + 16);
  v16bf out;
#pragma unroll
  for (int i = 0; i < 8; ++i) { out[i] = lo[i]; out[i + 8] = hi[i]; }
  return out;
}

__device__ __forceinline__ v8f wmma_bf16(v16bf a, v16bf b, v8f c) {
  return __builtin_amdgcn_wmma_f32_16x16x32_bf16(false, a, false, b, (short)0, c,
                                                 false, false);
}

// gfx1250 async DMA: global -> LDS, 16B per lane, tracked by ASYNCcnt
__device__ __forceinline__ void async_load_b128(bf16_t* lds_p, const bf16_t* g_p) {
  unsigned           loff = (unsigned)(unsigned long long)lds_p;
  unsigned long long ga   = (unsigned long long)g_p;
  asm volatile("global_load_async_to_lds_b128 %0, %1, off"
               :: "v"(loff), "v"(ga) : "memory");
}

// ---------------------------------------------------------------------------
// f32 -> bf16 convert, 8 elems/thread
// ---------------------------------------------------------------------------
__global__ void f32_to_bf16_kernel(const float* __restrict__ in,
                                   bf16_t* __restrict__ out, int n8) {
  int i = blockIdx.x * blockDim.x + threadIdx.x;
  if (i >= n8) return;
  const float4* p = (const float4*)in + (size_t)i * 2;
  float4 a = p[0], b = p[1];
  v8bf o;
  o[0] = (bf16_t)a.x; o[1] = (bf16_t)a.y; o[2] = (bf16_t)a.z; o[3] = (bf16_t)a.w;
  o[4] = (bf16_t)b.x; o[5] = (bf16_t)b.y; o[6] = (bf16_t)b.z; o[7] = (bf16_t)b.w;
  ((v8bf*)out)[i] = o;
}

// ---------------------------------------------------------------------------
// C[M,N] = A[M,K] @ W[N,K]^T + bias[N]; wave tile 32x64 (8 WMMA / k-step).
// sched_barrier pins: batch all 6 frag loads, then run the 8 WMMAs.
// ---------------------------------------------------------------------------
template <typename OutT>
__global__ __launch_bounds__(256) void gemm_bf16_wmma_kernel(
    const bf16_t* __restrict__ A, const bf16_t* __restrict__ W,
    const float* __restrict__ bias, OutT* __restrict__ C, int M, int N, int K) {
  int wave = threadIdx.x >> 5;
  int lane = threadIdx.x & 31;
  int m0 = blockIdx.x * 256 + wave * 32;
  int n0 = blockIdx.y * 64;

  v8f acc0[4] = {}, acc1[4] = {};
  for (int k = 0; k < K; k += 32) {
    v16bf b0 = load_b_frag(W + (size_t)(n0 +  0) * K + k, K);
    v16bf b1 = load_b_frag(W + (size_t)(n0 + 16) * K + k, K);
    v16bf b2 = load_b_frag(W + (size_t)(n0 + 32) * K + k, K);
    v16bf b3 = load_b_frag(W + (size_t)(n0 + 48) * K + k, K);
    v16bf a0 = load_a_frag(A + (size_t)m0 * K + k, K);
    v16bf a1 = load_a_frag(A + (size_t)(m0 + 16) * K + k, K);
    __builtin_amdgcn_sched_barrier(0);
    acc0[0] = wmma_bf16(a0, b0, acc0[0]);
    acc0[1] = wmma_bf16(a0, b1, acc0[1]);
    acc0[2] = wmma_bf16(a0, b2, acc0[2]);
    acc0[3] = wmma_bf16(a0, b3, acc0[3]);
    acc1[0] = wmma_bf16(a1, b0, acc1[0]);
    acc1[1] = wmma_bf16(a1, b1, acc1[1]);
    acc1[2] = wmma_bf16(a1, b2, acc1[2]);
    acc1[3] = wmma_bf16(a1, b3, acc1[3]);
  }

  int mbase = (lane >> 4) * 8;
  int ncol  = lane & 15;
#pragma unroll
  for (int t = 0; t < 4; ++t) {
    int n = n0 + t * 16 + ncol;
    float bv = bias[n];
#pragma unroll
    for (int r = 0; r < 8; ++r) {
      C[(size_t)(m0 + mbase + r) * N + n]      = (OutT)(acc0[t][r] + bv);
      C[(size_t)(m0 + 16 + mbase + r) * N + n] = (OutT)(acc1[t][r] + bv);
    }
  }
}

// ---------------------------------------------------------------------------
// V slice of qkv -> Vt[B,H,Dh,T] via LDS-tiled 64x64 transpose (coalesced)
// ---------------------------------------------------------------------------
__global__ __launch_bounds__(256) void v_transpose_kernel(
    const bf16_t* __restrict__ qkv, bf16_t* __restrict__ Vt) {
  __shared__ bf16_t tile[64][72];
  int bh = blockIdx.x;
  int b = bh >> 4, h = bh & 15;
  int t0 = blockIdx.y * 64;
  int tid = threadIdx.x;

  int row = tid >> 2;            // t index 0..63
  int cg  = (tid & 3) * 16;      // d group
  const bf16_t* src = qkv + (size_t)(b * TT + t0 + row) * E3 + 2 * EMBED + h * HDIM + cg;
  v8bf a0 = *(const v8bf*)(src);
  v8bf a1 = *(const v8bf*)(src + 8);
#pragma unroll
  for (int e = 0; e < 8; ++e) { tile[row][cg + e] = a0[e]; tile[row][cg + 8 + e] = a1[e]; }
  __syncthreads();

  int d  = tid >> 2;             // dh index 0..63
  int tg = (tid & 3) * 16;       // t group
  v8bf o0, o1;
#pragma unroll
  for (int j = 0; j < 8; ++j) { o0[j] = tile[tg + j][d]; o1[j] = tile[tg + 8 + j][d]; }
  bf16_t* dst = Vt + (size_t)(bh * HDIM + d) * TT + t0 + tg;
  *(v8bf*)(dst)     = o0;
  *(v8bf*)(dst + 8) = o1;
}

// ---------------------------------------------------------------------------
// Flash attention. grid (B*H, T/128); 8 waves, 16 queries/wave, 64-key chunks.
// K/V double-buffered in LDS via async global->LDS DMA (ASYNCcnt).
// sched_barrier after each frag-load batch -> distinct VGPRs, partial waits.
// ---------------------------------------------------------------------------
__global__ __launch_bounds__(256) void flash_attn_kernel(
    const bf16_t* __restrict__ qkv, const bf16_t* __restrict__ Vt,
    const float* __restrict__ mask, bf16_t* __restrict__ O) {
  __shared__ __align__(16) bf16_t Ksh[2][64 * 64];   // [key][dh]
  __shared__ __align__(16) bf16_t Vsh[2][64 * 64];   // [dh][key]
  __shared__ __align__(16) bf16_t Psh[8][16 * 64];   // per-wave P tile

  int bh = blockIdx.x;
  int b = bh >> 4, h = bh & 15;
  int wave = threadIdx.x >> 5;
  int lane = threadIdx.x & 31;
  int tid  = threadIdx.x;
  int q0 = blockIdx.y * 128 + wave * 16;

  const bf16_t* Qg = qkv + (size_t)(b * TT) * E3 + h * HDIM;          // stride E3
  const bf16_t* Kg = qkv + (size_t)(b * TT) * E3 + EMBED + h * HDIM;  // stride E3
  const bf16_t* Vg = Vt + (size_t)bh * HDIM * TT;                     // stride TT
  const float*  mb = mask + (size_t)b * TT;

  v16bf qf0 = load_a_frag(Qg + (size_t)q0 * E3 + 0,  E3);
  v16bf qf1 = load_a_frag(Qg + (size_t)q0 * E3 + 32, E3);

  v8f o[4] = {};
  float mrow[8], lrow[8];
#pragma unroll
  for (int r = 0; r < 8; ++r) { mrow[r] = -1e30f; lrow[r] = 0.f; }

  bf16_t* pw = Psh[wave];
  int ncol = lane & 15;
  int half = lane >> 4;
  const float scale = 0.125f;   // 1/sqrt(Dh)

  // stage one 64-key chunk: 512 x 16B for K + same for V, 4 DMAs/thread
  auto stage = [&](int kb, int buf) {
#pragma unroll
    for (int j = 0; j < 2; ++j) {
      int i = tid + 256 * j;          // 0..511
      int row = i >> 3, seg = i & 7;  // seg: 8 bf16 = 16B
      async_load_b128(&Ksh[buf][i * 8], Kg + (size_t)(kb + row) * E3 + seg * 8);
      async_load_b128(&Vsh[buf][i * 8], Vg + (size_t)row * TT + kb + seg * 8);
    }
  };

  stage(0, 0);
  for (int kb = 0; kb < TT; kb += 64) {
    int cur = (kb >> 6) & 1;
    const bf16_t* Kc = Ksh[cur];
    const bf16_t* Vc = Vsh[cur];

    asm volatile("s_wait_asynccnt 0" ::: "memory");
    __syncthreads();                      // chunk `cur` visible to all waves
    if (kb + 64 < TT) stage(kb + 64, cur ^ 1);

    // mask values for this chunk's 64 key columns (hide under S WMMAs)
    float mv[4];
#pragma unroll
    for (int t = 0; t < 4; ++t) mv[t] = mb[kb + t * 16 + ncol];

    // ---- S = Q @ K^T : batched loads | sched_barrier | WMMA group
    v8f s[4] = {};
    {
      v16bf k0 = load_b_frag(Kc +  0 * 64, 64);
      v16bf k1 = load_b_frag(Kc + 16 * 64, 64);
      v16bf k2 = load_b_frag(Kc + 32 * 64, 64);
      v16bf k3 = load_b_frag(Kc + 48 * 64, 64);
      __builtin_amdgcn_sched_barrier(0);
      s[0] = wmma_bf16(qf0, k0, s[0]);
      s[1] = wmma_bf16(qf0, k1, s[1]);
      s[2] = wmma_bf16(qf0, k2, s[2]);
      s[3] = wmma_bf16(qf0, k3, s[3]);
      v16bf k4 = load_b_frag(Kc +  0 * 64 + 32, 64);
      v16bf k5 = load_b_frag(Kc + 16 * 64 + 32, 64);
      v16bf k6 = load_b_frag(Kc + 32 * 64 + 32, 64);
      v16bf k7 = load_b_frag(Kc + 48 * 64 + 32, 64);
      __builtin_amdgcn_sched_barrier(0);
      s[0] = wmma_bf16(qf1, k4, s[0]);
      s[1] = wmma_bf16(qf1, k5, s[1]);
      s[2] = wmma_bf16(qf1, k6, s[2]);
      s[3] = wmma_bf16(qf1, k7, s[3]);
    }

    // ---- online softmax: row-MAX butterfly only; row-sum stays lane-partial
    float nm[8];
#pragma unroll
    for (int r = 0; r < 8; ++r) {
      float c = -1e30f;
#pragma unroll
      for (int t = 0; t < 4; ++t) {
        s[t][r] = s[t][r] * scale + mv[t];
        c = fmaxf(c, s[t][r]);
      }
#pragma unroll
      for (int off = 1; off < 16; off <<= 1) c = fmaxf(c, __shfl_xor(c, off, 32));
      nm[r] = fmaxf(mrow[r], c);
    }
#pragma unroll
    for (int r = 0; r < 8; ++r) {
      float alpha = __expf(mrow[r] - nm[r]);
      mrow[r] = nm[r];
      lrow[r] *= alpha;
#pragma unroll
      for (int t = 0; t < 4; ++t) o[t][r] *= alpha;
      int rowi = half * 8 + r;
#pragma unroll
      for (int t = 0; t < 4; ++t) {
        float p = __expf(s[t][r] - nm[r]);
        lrow[r] += p;
        pw[rowi * 64 + t * 16 + ncol] = (bf16_t)p;
      }
    }
    asm volatile("s_wait_dscnt 0" ::: "memory");  // P stores -> cross-lane reads

    // ---- O += P @ V : batched loads | sched_barrier | WMMA group
    {
      v16bf pf0 = load_a_frag(pw + 0,  64);
      v16bf v0 = load_b_frag(Vc +  0 * 64, 64);
      v16bf v1 = load_b_frag(Vc + 16 * 64, 64);
      v16bf v2 = load_b_frag(Vc + 32 * 64, 64);
      v16bf v3 = load_b_frag(Vc + 48 * 64, 64);
      __builtin_amdgcn_sched_barrier(0);
      o[0] = wmma_bf16(pf0, v0, o[0]);
      o[1] = wmma_bf16(pf0, v1, o[1]);
      o[2] = wmma_bf16(pf0, v2, o[2]);
      o[3] = wmma_bf16(pf0, v3, o[3]);
      v16bf pf1 = load_a_frag(pw + 32, 64);
      v16bf v4 = load_b_frag(Vc +  0 * 64 + 32, 64);
      v16bf v5 = load_b_frag(Vc + 16 * 64 + 32, 64);
      v16bf v6 = load_b_frag(Vc + 32 * 64 + 32, 64);
      v16bf v7 = load_b_frag(Vc + 48 * 64 + 32, 64);
      __builtin_amdgcn_sched_barrier(0);
      o[0] = wmma_bf16(pf1, v4, o[0]);
      o[1] = wmma_bf16(pf1, v5, o[1]);
      o[2] = wmma_bf16(pf1, v6, o[2]);
      o[3] = wmma_bf16(pf1, v7, o[3]);
    }
  }

  // ---- final row-sum butterfly, normalize, store [B,T,E] bf16
#pragma unroll
  for (int r = 0; r < 8; ++r) {
    float s = lrow[r];
#pragma unroll
    for (int off = 1; off < 16; off <<= 1) s += __shfl_xor(s, off, 32);
    lrow[r] = 1.f / s;
  }
#pragma unroll
  for (int r = 0; r < 8; ++r) {
    int trow = q0 + half * 8 + r;
#pragma unroll
    for (int t = 0; t < 4; ++t) {
      int col = h * HDIM + t * 16 + ncol;
      O[(size_t)(b * TT + trow) * EMBED + col] = (bf16_t)(o[t][r] * lrow[r]);
    }
  }
}

// ---------------------------------------------------------------------------
extern "C" void kernel_launch(void* const* d_in, const int* in_sizes, int n_in,
                              void* d_out, int out_size, void* d_ws,
                              size_t ws_size, hipStream_t stream) {
  const float* x    = (const float*)d_in[0];   // [B,T,E]
  const float* mask = (const float*)d_in[1];   // [B,T]
  const float* win  = (const float*)d_in[2];   // [3E,E]
  const float* bin  = (const float*)d_in[3];   // [3E]
  const float* wout = (const float*)d_in[4];   // [E,E]
  const float* bout = (const float*)d_in[5];   // [E]
  float* out = (float*)d_out;                  // [B,T,E] f32

  char* ws = (char*)d_ws;
  bf16_t* xb   = (bf16_t*)ws;  ws += (size_t)NT * EMBED * 2;
  bf16_t* wib  = (bf16_t*)ws;  ws += (size_t)E3 * EMBED * 2;
  bf16_t* wob  = (bf16_t*)ws;  ws += (size_t)EMBED * EMBED * 2;
  bf16_t* qkvb = (bf16_t*)ws;  ws += (size_t)NT * E3 * 2;
  bf16_t* Vt   = (bf16_t*)ws;  ws += (size_t)NT * EMBED * 2;
  bf16_t* Ob   = (bf16_t*)ws;  ws += (size_t)NT * EMBED * 2;

  int n1 = NT * EMBED / 8;
  f32_to_bf16_kernel<<<(n1 + 255) / 256, 256, 0, stream>>>(x, xb, n1);
  int n2 = E3 * EMBED / 8;
  f32_to_bf16_kernel<<<(n2 + 255) / 256, 256, 0, stream>>>(win, wib, n2);
  int n3 = EMBED * EMBED / 8;
  f32_to_bf16_kernel<<<(n3 + 255) / 256, 256, 0, stream>>>(wout, wob, n3);

  // QKV projection: M=8192, N=3072, K=1024
  gemm_bf16_wmma_kernel<bf16_t><<<dim3(NT / 256, E3 / 64), 256, 0, stream>>>(
      xb, wib, bin, qkvb, NT, E3, EMBED);

  // only V needs repacking (transpose); Q/K are read in place by flash
  v_transpose_kernel<<<dim3(BB * NHEADS, TT / 64), 256, 0, stream>>>(qkvb, Vt);

  // fused flash attention
  flash_attn_kernel<<<dim3(BB * NHEADS, TT / 128), 256, 0, stream>>>(
      qkvb, Vt, mask, Ob);

  // output projection: M=8192, N=1024, K=1024, f32 out
  gemm_bf16_wmma_kernel<float><<<dim3(NT / 256, EMBED / 64), 256, 0, stream>>>(
      Ob, wob, bout, out, NT, EMBED, EMBED);
}
